// HyperbolicAttention_10574209483387
// MI455X (gfx1250) — compile-verified, matrix operand order checked by reading, not run on hardware
//
#include <hip/hip_runtime.h>
#include <math.h>

#define HEADS    4
#define HEAD_DIM 32
#define DDIM     128
#define SCALE_F  0.17677669529663687f   /* 32^-0.5 */
#define EPS_F    1e-7f

typedef __attribute__((ext_vector_type(16))) _Float16 v16h;
typedef __attribute__((ext_vector_type(8)))  float    v8f;
typedef __attribute__((ext_vector_type(2)))  _Float16 h2;

/* ------------------------------------------------------------------ */
/* 1. Lorentz log map at origin -> f16 tangent features               */
/* ------------------------------------------------------------------ */
__global__ void log_origin_f16_kernel(const float* __restrict__ x,
                                      _Float16* __restrict__ xtan_h,
                                      int n_nodes) {
    int n = blockIdx.x * blockDim.x + threadIdx.x;
    if (n >= n_nodes) return;
    const float* xr = x + (size_t)n * (DDIM + 1);
    float x0 = xr[0];
    float nrm2 = 0.0f;
#pragma unroll 4
    for (int d = 0; d < DDIM; ++d) {
        float v = xr[1 + d];
        nrm2 += v * v;
    }
    float norm = sqrtf(nrm2 + EPS_F);
    float arg  = fmaxf(x0, 1.0f + EPS_F);           /* sc = sqrt(CURV) = 1 */
    float dist = acoshf(arg);
    float f    = dist / norm;
    _Float16* o = xtan_h + (size_t)n * DDIM;
#pragma unroll 4
    for (int d = 0; d < DDIM; ++d)
        o[d] = (_Float16)(f * xr[1 + d]);
}

/* ------------------------------------------------------------------ */
/* 2. generic f32 -> f16 convert                                      */
/* ------------------------------------------------------------------ */
__global__ void f32_to_f16_kernel(const float* __restrict__ src,
                                  _Float16* __restrict__ dst, int n) {
    int i = blockIdx.x * blockDim.x + threadIdx.x;
    if (i < n) dst[i] = (_Float16)src[i];
}

/* ------------------------------------------------------------------ */
/* 3. init: agg = 0, denom = 0, m = -inf                              */
/* ------------------------------------------------------------------ */
__global__ void init_kernel(float* __restrict__ agg,
                            float* __restrict__ denom,
                            unsigned int* __restrict__ mbits,
                            int n_nodes) {
    int i = blockIdx.x * blockDim.x + threadIdx.x;
    if (i < n_nodes * DDIM) agg[i] = 0.0f;
    if (i < n_nodes * HEADS) {
        denom[i] = 0.0f;
        mbits[i] = 0xFF800000u;                      /* -inf */
    }
}

/* ------------------------------------------------------------------ */
/* 4. WMMA GEMM:  Y[M,128] = Xh[M,128] @ Wh[128,128]^T  (f16 -> f32)  */
/*    one wave = 16-row block x all 8 column tiles                    */
/*    Layouts per CDNA5 ISA 7.12.2 (16-bit A 16x32, f32 C/D 16x16).   */
/* ------------------------------------------------------------------ */
__global__ void gemm_xwT_wmma(const _Float16* __restrict__ X,
                              const _Float16* __restrict__ W,
                              float* __restrict__ Y, int nrows) {
    const int wave = blockIdx.x * (blockDim.x >> 5) + (threadIdx.x >> 5);
    const int m0   = wave << 4;
    if (m0 >= nrows) return;                 /* wave-uniform: EXEC stays full */
    const int lane = threadIdx.x & 31;
    const int li   = lane & 15;
    const int hi   = lane >> 4;

    v8f acc[8];
#pragma unroll
    for (int i = 0; i < 8; ++i) acc[i] = (v8f){0,0,0,0,0,0,0,0};

    const _Float16* xrow = X + (size_t)(m0 + li) * DDIM;

#pragma unroll
    for (int k0 = 0; k0 < DDIM; k0 += 32) {
        /* A tile 16x32: lane half hi picks K in [0,8)+[16,24) or [8,16)+[24,32) */
        v16h a;
#pragma unroll
        for (int v = 0; v < 8; ++v) {
            const int kk = k0 + 8 * hi + ((v < 4) ? (2 * v) : (16 + 2 * (v - 4)));
            h2 p = *(const h2*)(xrow + kk);
            a[2 * v]     = p.x;
            a[2 * v + 1] = p.y;
        }
#pragma unroll
        for (int nt = 0; nt < 8; ++nt) {
            /* B tile 32x16 of W^T: B[k][n] = W[n][k] (row-major W, tiny, L1-hot) */
            const _Float16* wrow = W + (size_t)(nt * 16 + li) * DDIM;
            v16h b;
#pragma unroll
            for (int v = 0; v < 8; ++v) {
                const int kk = k0 + 8 * hi + ((v < 4) ? (2 * v) : (16 + 2 * (v - 4)));
                h2 p = *(const h2*)(wrow + kk);
                b[2 * v]     = p.x;
                b[2 * v + 1] = p.y;
            }
            acc[nt] = __builtin_amdgcn_wmma_f32_16x16x32_f16(
                false, a, false, b, (short)0, acc[nt], false, false);
        }
    }
    /* f32 C/D layout: VGPR r, lane -> (M = m0 + r + 8*hi, N = nt*16 + li) */
#pragma unroll
    for (int nt = 0; nt < 8; ++nt)
#pragma unroll
        for (int r = 0; r < 8; ++r)
            Y[(size_t)(m0 + r + 8 * hi) * DDIM + nt * 16 + li] = acc[nt][r];
}

/* ------------------------------------------------------------------ */
/* 5. edge pass 1: scores + segment-max (atomic float-max via punning)*/
/* ------------------------------------------------------------------ */
__device__ __forceinline__ void atomicMaxF(float* addr, float val) {
    if (val >= 0.0f) atomicMax((int*)addr, __float_as_int(val));
    else             atomicMin((unsigned int*)addr, __float_as_uint(val));
}

__global__ void edge_score_max_kernel(const long long* __restrict__ ei,
                                      const float* __restrict__ Q,
                                      const float* __restrict__ K,
                                      float* __restrict__ scores,
                                      float* __restrict__ m,
                                      int n_edges) {
    int t = blockIdx.x * blockDim.x + threadIdx.x;
    if (t >= n_edges * HEADS) return;
    int e = t >> 2, h = t & 3;
    long long row = ei[e];
    long long col = ei[(size_t)n_edges + e];
    const float4* q = (const float4*)(Q + (size_t)col * DDIM + h * HEAD_DIM);
    const float4* k = (const float4*)(K + (size_t)row * DDIM + h * HEAD_DIM);
    float s = 0.0f;
#pragma unroll
    for (int i = 0; i < HEAD_DIM / 4; ++i) {
        float4 a = q[i], b = k[i];
        s += a.x * b.x + a.y * b.y + a.z * b.z + a.w * b.w;
    }
    s *= SCALE_F;
    scores[t] = s;
    atomicMaxF(&m[(size_t)col * HEADS + h], s);
}

/* ------------------------------------------------------------------ */
/* 6. edge pass 2: e = exp(s - m[col]); denom += e                    */
/* ------------------------------------------------------------------ */
__global__ void edge_exp_sum_kernel(const long long* __restrict__ ei,
                                    float* __restrict__ scores,
                                    const float* __restrict__ m,
                                    float* __restrict__ denom,
                                    int n_edges) {
    int t = blockIdx.x * blockDim.x + threadIdx.x;
    if (t >= n_edges * HEADS) return;
    int e = t >> 2, h = t & 3;
    long long col = ei[(size_t)n_edges + e];
    float ev = expf(scores[t] - m[(size_t)col * HEADS + h]);
    scores[t] = ev;
    atomicAdd(&denom[(size_t)col * HEADS + h], ev);
}

/* ------------------------------------------------------------------ */
/* 7. edge pass 3: agg[col] += (e/denom[col]) * V[row]                */
/* ------------------------------------------------------------------ */
__global__ void edge_agg_kernel(const long long* __restrict__ ei,
                                const float* __restrict__ scores,
                                const float* __restrict__ denom,
                                const float* __restrict__ V,
                                float* __restrict__ agg,
                                int n_edges) {
    int t = blockIdx.x * blockDim.x + threadIdx.x;
    if (t >= n_edges * HEADS) return;
    int e = t >> 2, h = t & 3;
    long long row = ei[e];
    long long col = ei[(size_t)n_edges + e];
    float att = scores[t] / denom[(size_t)col * HEADS + h];
    const float* v = V   + (size_t)row * DDIM + h * HEAD_DIM;
    float*       o = agg + (size_t)col * DDIM + h * HEAD_DIM;
#pragma unroll 4
    for (int d = 0; d < HEAD_DIM; ++d)
        atomicAdd(&o[d], att * v[d]);
}

/* ------------------------------------------------------------------ */
/* 8. exp map at origin (+ output bias) -> d_out (N x 129 f32)        */
/* ------------------------------------------------------------------ */
__global__ void exp_origin_kernel(const float* __restrict__ out_tan,
                                  const float* __restrict__ bo,
                                  float* __restrict__ out,
                                  int n_nodes) {
    int n = blockIdx.x * blockDim.x + threadIdx.x;
    if (n >= n_nodes) return;
    const float* v = out_tan + (size_t)n * DDIM;
    float r2 = 0.0f;
#pragma unroll 4
    for (int d = 0; d < DDIM; ++d) {
        float t = v[d] + bo[d];
        r2 += t * t;
    }
    float r   = sqrtf(r2 + EPS_F);
    float arg = r;                                   /* sc = 1 */
    float x0  = coshf(arg);
    float fac = sinhf(arg) / arg;
    float* o = out + (size_t)n * (DDIM + 1);
    o[0] = x0;
#pragma unroll 4
    for (int d = 0; d < DDIM; ++d)
        o[1 + d] = fac * (v[d] + bo[d]);
}

/* ------------------------------------------------------------------ */
/* host launcher                                                      */
/* ------------------------------------------------------------------ */
extern "C" void kernel_launch(void* const* d_in, const int* in_sizes, int n_in,
                              void* d_out, int out_size, void* d_ws, size_t ws_size,
                              hipStream_t stream) {
    const float*     x  = (const float*)d_in[0];
    const long long* ei = (const long long*)d_in[1];   /* int64 edge_index */
    const float*     Wq = (const float*)d_in[2];
    const float*     Wk = (const float*)d_in[3];
    const float*     Wv = (const float*)d_in[4];
    const float*     Wo = (const float*)d_in[5];
    const float*     bo = (const float*)d_in[6];
    float*           out = (float*)d_out;

    const int N = in_sizes[0] / (DDIM + 1);            /* 50000 */
    const int E = in_sizes[1] / 2;                     /* 800000 */

    /* -------- workspace carve-up (256 B aligned) -------- */
    char*  base = (char*)d_ws;
    size_t off  = 0;
    auto carve = [&](size_t bytes) -> char* {
        char* p = base + off;
        off = (off + bytes + 255) & ~(size_t)255;
        return p;
    };
    _Float16* xtan_h = (_Float16*)carve((size_t)N * DDIM * 2);
    _Float16* Wq_h   = (_Float16*)carve((size_t)DDIM * DDIM * 2);
    _Float16* Wk_h   = (_Float16*)carve((size_t)DDIM * DDIM * 2);
    _Float16* Wv_h   = (_Float16*)carve((size_t)DDIM * DDIM * 2);
    _Float16* Wo_h   = (_Float16*)carve((size_t)DDIM * DDIM * 2);
    float*    Qf     = (float*)carve((size_t)N * DDIM * 4);
    float*    Kf     = (float*)carve((size_t)N * DDIM * 4);
    float*    Vf     = (float*)carve((size_t)N * DDIM * 4);
    float*    scores = (float*)carve((size_t)E * HEADS * 4);
    float*    mmax   = (float*)carve((size_t)N * HEADS * 4);
    float*    denom  = (float*)carve((size_t)N * HEADS * 4);
    float*    agg    = (float*)carve((size_t)N * DDIM * 4);
    _Float16* agg_h  = (_Float16*)carve((size_t)N * DDIM * 2);
    float*    otan   = (float*)carve((size_t)N * DDIM * 4);
    (void)ws_size; (void)n_in; (void)out_size;

    const int TB = 256;

    /* 1. log map -> f16 features */
    log_origin_f16_kernel<<<(N + TB - 1) / TB, TB, 0, stream>>>(x, xtan_h, N);

    /* 2. weights -> f16 */
    const int WN = DDIM * DDIM;
    f32_to_f16_kernel<<<(WN + TB - 1) / TB, TB, 0, stream>>>(Wq, Wq_h, WN);
    f32_to_f16_kernel<<<(WN + TB - 1) / TB, TB, 0, stream>>>(Wk, Wk_h, WN);
    f32_to_f16_kernel<<<(WN + TB - 1) / TB, TB, 0, stream>>>(Wv, Wv_h, WN);
    f32_to_f16_kernel<<<(WN + TB - 1) / TB, TB, 0, stream>>>(Wo, Wo_h, WN);

    /* 3. init accumulators */
    init_kernel<<<((size_t)N * DDIM + TB - 1) / TB, TB, 0, stream>>>(
        agg, denom, (unsigned int*)mmax, N);

    /* 4. Q/K/V projections via WMMA (N divisible by 16) */
    const int row_blocks = N / 16;                     /* 3125 waves */
    const int waves_pb   = TB / 32;                    /* 8 */
    const int ggrid      = (row_blocks + waves_pb - 1) / waves_pb;
    gemm_xwT_wmma<<<ggrid, TB, 0, stream>>>(xtan_h, Wq_h, Qf, N);
    gemm_xwT_wmma<<<ggrid, TB, 0, stream>>>(xtan_h, Wk_h, Kf, N);
    gemm_xwT_wmma<<<ggrid, TB, 0, stream>>>(xtan_h, Wv_h, Vf, N);

    /* 5-7. edge phase */
    const int EH = E * HEADS;
    edge_score_max_kernel<<<(EH + TB - 1) / TB, TB, 0, stream>>>(ei, Qf, Kf, scores, mmax, E);
    edge_exp_sum_kernel  <<<(EH + TB - 1) / TB, TB, 0, stream>>>(ei, scores, mmax, denom, E);
    edge_agg_kernel      <<<(EH + TB - 1) / TB, TB, 0, stream>>>(ei, scores, denom, Vf, agg, E);

    /* 8. output projection via WMMA */
    const int AN = N * DDIM;
    f32_to_f16_kernel<<<(AN + TB - 1) / TB, TB, 0, stream>>>(agg, agg_h, AN);
    gemm_xwT_wmma<<<ggrid, TB, 0, stream>>>(agg_h, Wo_h, otan, N);

    /* 9. exp map + bias -> final output */
    exp_origin_kernel<<<(N + TB - 1) / TB, TB, 0, stream>>>(otan, bo, out, N);
}
